// ConvAttn_22917945491862
// MI455X (gfx1250) — matrix operand order repaired and checked by reading, not compiled, not run on hardware
//
#include <hip/hip_runtime.h>
#include <cstddef>

// ---------- types ----------
typedef __attribute__((ext_vector_type(16))) __bf16 v16bf;
typedef __attribute__((ext_vector_type(8)))  float  v8f;
typedef __attribute__((ext_vector_type(4)))  unsigned int v4u;

union ABu { v16bf v; v4u q[2]; };

__device__ __forceinline__ unsigned short f2bf(float f) {
  unsigned int u = __float_as_uint(f);
  u = (u + 0x7FFFu + ((u >> 16) & 1u)) >> 16;
  return (unsigned short)u;
}

// Dims: B=8, C=64, S=256, F=256
// NHWC bf16 activations: [b][h][w][c], c innermost (64)
// B-fragment pack: per 32-K chunk: [nc(4)][lane(32)][16 halfs], lane: col n=lane%16, k=(lane>>4)*16+j

// ---------- weight packing ----------
__global__ __launch_bounds__(256) void k_pack_res(const float* __restrict__ W,
                                                  unsigned short* __restrict__ frag) {
  int idx = blockIdx.x * 256 + threadIdx.x;           // 9*2*4*32*16 = 36864
  if (idx >= 36864) return;
  const int j = idx & 15, lane = (idx >> 4) & 31, nc = (idx >> 9) & 3;
  const int kc = (idx >> 11) & 1, tap = idx >> 12;
  const int co = nc * 16 + (lane & 15);
  const int ci = kc * 32 + (lane >> 4) * 16 + j;
  const int kh = tap / 3, kw = tap % 3;
  frag[idx] = f2bf(W[((co * 64 + ci) * 3 + kh) * 3 + kw]);
}

__global__ __launch_bounds__(256) void k_pack_wide(const float* __restrict__ W,
                                                   const float* __restrict__ g,
                                                   unsigned short* __restrict__ frag, int mode) {
  int idx = blockIdx.x * 256 + threadIdx.x;           // 3*256*2*4*32*16 = 3145728
  if (idx >= 3145728) return;
  const int j = idx & 15, lane = (idx >> 4) & 31, nc = (idx >> 9) & 3;
  const int kc = (idx >> 11) & 1, pos = (idx >> 12) & 255, tap = idx >> 20;
  const int co = nc * 16 + (lane & 15);
  const int ci = kc * 32 + (lane >> 4) * 16 + j;
  float w;
  if (mode == 0) w = W[((co * 64 + ci) * 3 + tap) * 256 + pos];   // (co,ci,3,256)
  else           w = W[((co * 64 + ci) * 256 + pos) * 3 + tap];   // (co,ci,256,3)
  frag[idx] = f2bf(g[ci] * w);
}

__global__ __launch_bounds__(256) void k_fold_bias(const float* __restrict__ W,
                                                   const float* __restrict__ beta,
                                                   const float* __restrict__ cb,
                                                   float* __restrict__ bo) {
  const int co = blockIdx.x;                          // 64 blocks
  float s = 0.f;
  for (int i = threadIdx.x; i < 49152; i += 256) {
    const int ci = i / 768;                           // per-ci block = 3*256 in both shapes
    s += beta[ci] * W[(size_t)co * 49152 + i];
  }
  __shared__ float rs[256];
  rs[threadIdx.x] = s; __syncthreads();
  for (int off = 128; off > 0; off >>= 1) {
    if (threadIdx.x < off) rs[threadIdx.x] += rs[threadIdx.x + off];
    __syncthreads();
  }
  if (threadIdx.x == 0) bo[co] = cb[co] + rs[0];
}

// ---------- BN statistics ----------
__global__ __launch_bounds__(256) void k_stats_nchw(const float* __restrict__ x,
                                                    float* __restrict__ mean,
                                                    float* __restrict__ rstd) {
  const int c = blockIdx.x;                           // 64 blocks
  float s = 0.f, s2 = 0.f;
  for (int b = 0; b < 8; ++b) {
    const float* p = x + ((size_t)(b * 64 + c)) * 65536;
    for (int i = threadIdx.x; i < 65536; i += 256) { float v = p[i]; s += v; s2 += v * v; }
  }
  __shared__ float rs[256], rq[256];
  rs[threadIdx.x] = s; rq[threadIdx.x] = s2; __syncthreads();
  for (int off = 128; off > 0; off >>= 1) {
    if (threadIdx.x < off) { rs[threadIdx.x] += rs[threadIdx.x + off]; rq[threadIdx.x] += rq[threadIdx.x + off]; }
    __syncthreads();
  }
  if (threadIdx.x == 0) {
    const float N = 524288.f;
    float mu = rs[0] / N;
    mean[c] = mu;
    rstd[c] = rsqrtf(rq[0] / N - mu * mu + 1e-5f);
  }
}

__global__ __launch_bounds__(256) void k_stats_nhwc_part(const float* __restrict__ x,
                                                         float* __restrict__ part) {
  const int c = threadIdx.x & 63, sub = threadIdx.x >> 6;  // 256 blocks
  const size_t base = (size_t)blockIdx.x * 2048;
  float s = 0.f, s2 = 0.f;
  for (int i = sub; i < 2048; i += 4) { float v = x[(base + i) * 64 + c]; s += v; s2 += v * v; }
  __shared__ float rs[256], rq[256];
  rs[threadIdx.x] = s; rq[threadIdx.x] = s2; __syncthreads();
  if (sub == 0) {
    float ts = rs[c] + rs[c + 64] + rs[c + 128] + rs[c + 192];
    float t2 = rq[c] + rq[c + 64] + rq[c + 128] + rq[c + 192];
    part[((size_t)blockIdx.x * 64 + c) * 2]     = ts;
    part[((size_t)blockIdx.x * 64 + c) * 2 + 1] = t2;
  }
}

__global__ void k_stats_nhwc_fin(const float* __restrict__ part,
                                 float* __restrict__ mean, float* __restrict__ rstd) {
  const int c = threadIdx.x;                          // 1 block, 64 threads
  float s = 0.f, s2 = 0.f;
  for (int g = 0; g < 256; ++g) {
    s  += part[((size_t)g * 64 + c) * 2];
    s2 += part[((size_t)g * 64 + c) * 2 + 1];
  }
  const float N = 524288.f;
  float mu = s / N;
  mean[c] = mu;
  rstd[c] = rsqrtf(s2 / N - mu * mu + 1e-5f);
}

// ---------- elementwise / layout ----------
// NCHW f32 -> NHWC bf16, with optional affine+relu (LDS transpose for coalescing)
__global__ __launch_bounds__(256) void k_bn_nchw2nhwc(const float* __restrict__ x,
    const float* __restrict__ mean, const float* __restrict__ rstd,
    const float* __restrict__ g, const float* __restrict__ bt,
    unsigned short* __restrict__ out, int affine_relu) {
  const int h = blockIdx.x, b = blockIdx.y;
  __shared__ unsigned short tile[16384];              // 32 KB
  for (int idx = threadIdx.x; idx < 16384; idx += 256) {
    const int c = idx >> 8, w = idx & 255;
    float v = x[((size_t)(b * 64 + c) * 256 + h) * 256 + w];
    v = (v - mean[c]) * rstd[c];
    if (affine_relu) { v = fmaxf(v * g[c] + bt[c], 0.f); }
    tile[w * 64 + c] = f2bf(v);
  }
  __syncthreads();
  unsigned short* op = out + (size_t)(b * 256 + h) * 16384;
  for (int idx = threadIdx.x; idx < 16384; idx += 256) op[idx] = tile[idx];
}

// NHWC f32 -> NHWC bf16 BN+ReLU (pure elementwise)
__global__ __launch_bounds__(256) void k_bnrelu_nhwc(const float* __restrict__ x,
    const float* __restrict__ mean, const float* __restrict__ rstd,
    const float* __restrict__ g, const float* __restrict__ bt,
    unsigned short* __restrict__ out) {
  for (size_t i = (size_t)blockIdx.x * 256 + threadIdx.x; i < 33554432ull;
       i += (size_t)gridDim.x * 256) {
    const int c = (int)(i & 63);
    float v = x[i];
    v = fmaxf((v - mean[c]) * rstd[c] * g[c] + bt[c], 0.f);
    out[i] = f2bf(v);
  }
}

// x_new(NCHW) = x(NCHW) + v(NHWC)
__global__ __launch_bounds__(256) void k_resadd(const float* __restrict__ x,
                                                const float* __restrict__ vnhwc,
                                                float* __restrict__ xn) {
  const int h = blockIdx.x, b = blockIdx.y;
  __shared__ float tile[8192];                        // 32 KB
  const size_t ib = (size_t)(b * 256 + h) * 16384;
  for (int seg = 0; seg < 2; ++seg) {
    __syncthreads();
    for (int idx = threadIdx.x; idx < 8192; idx += 256)
      tile[idx] = vnhwc[ib + seg * 8192 + idx];       // [w_local][c]
    __syncthreads();
    for (int idx = threadIdx.x; idx < 8192; idx += 256) {
      const int c = idx >> 7, wl = idx & 127;
      const size_t o = ((size_t)(b * 64 + c) * 256 + h) * 256 + (seg * 128 + wl);
      xn[o] = x[o] + tile[wl * 64 + c];
    }
  }
}

// ---------- 3x3 conv, implicit GEMM, bf16 WMMA ----------
// act NHWC bf16, wfrag [9][2][4][32][16] staged once into 72KB dynamic LDS
// (all 8 waves of the block consume identical B fragments -> 8x L2 traffic cut)
__global__ __launch_bounds__(256) void k_conv3x3(const unsigned short* __restrict__ act,
                                                 const unsigned short* __restrict__ wfrag,
                                                 const float* __restrict__ bias,
                                                 float* __restrict__ out) {
  extern __shared__ unsigned short wlds[];            // 36864 halfs = 72 KB
  const int t = threadIdx.x;
  // cooperative stage of the full packed weight set
  for (int u = 0; u < 9; ++u) {
    const int linear = u * 4096 + t * 16;
    *(v4u*)(wlds + linear)     = *(const v4u*)(wfrag + linear);
    *(v4u*)(wlds + linear + 8) = *(const v4u*)(wfrag + linear + 8);
  }
  __syncthreads();

  const int lane = t & 31;
  const int wave = t >> 5;
  const int wt = blockIdx.x * 8 + wave;               // w-tile 0..15
  const int h = blockIdx.y;
  const int b = blockIdx.z;
  const int m = lane & 15;
  const int half = lane >> 4;
  const int w0 = wt * 16 + m;                         // M (spatial w) of this lane
  const v4u z4 = {0u, 0u, 0u, 0u};
  v8f acc0 = {}, acc1 = {}, acc2 = {}, acc3 = {};
  for (int tap = 0; tap < 9; ++tap) {
    const int kh = tap / 3, kw = tap % 3;
    const int hi = h + kh - 1;
    const int wi = w0 + kw - 1;
    const bool inb = (hi >= 0) && (hi < 256) && (wi >= 0) && (wi < 256);
    const size_t abase = inb ? ((size_t)((b * 256 + hi) * 256 + wi)) * 64 : 0;
    for (int kc = 0; kc < 2; ++kc) {
      ABu a;
      if (inb) {
        a.q[0] = *(const v4u*)(act + abase + kc * 32 + half * 16);
        a.q[1] = *(const v4u*)(act + abase + kc * 32 + half * 16 + 8);
      } else { a.q[0] = z4; a.q[1] = z4; }
      const unsigned short* wb = wlds + (tap * 2 + kc) * 2048 + lane * 16;
      ABu b0, b1, b2, b3;
      b0.q[0] = *(const v4u*)(wb);        b0.q[1] = *(const v4u*)(wb + 8);
      b1.q[0] = *(const v4u*)(wb + 512);  b1.q[1] = *(const v4u*)(wb + 520);
      b2.q[0] = *(const v4u*)(wb + 1024); b2.q[1] = *(const v4u*)(wb + 1032);
      b3.q[0] = *(const v4u*)(wb + 1536); b3.q[1] = *(const v4u*)(wb + 1544);
      acc0 = __builtin_amdgcn_wmma_f32_16x16x32_bf16(false, a.v, false, b0.v, (short)0, acc0, false, false);
      acc1 = __builtin_amdgcn_wmma_f32_16x16x32_bf16(false, a.v, false, b1.v, (short)0, acc1, false, false);
      acc2 = __builtin_amdgcn_wmma_f32_16x16x32_bf16(false, a.v, false, b2.v, (short)0, acc2, false, false);
      acc3 = __builtin_amdgcn_wmma_f32_16x16x32_bf16(false, a.v, false, b3.v, (short)0, acc3, false, false);
    }
  }
  const int n = lane & 15;
  const size_t ob = (size_t)(b * 256 + h) * 256;
  v8f accs[4] = {acc0, acc1, acc2, acc3};
  for (int nc = 0; nc < 4; ++nc) {
    const float bv = bias[nc * 16 + n];
    for (int r = 0; r < 8; ++r) {
      const int wo = wt * 16 + r + 8 * half;
      out[(ob + wo) * 64 + nc * 16 + n] = accs[nc][r] + bv;
    }
  }
}

// ---------- wide conv (seq: kernel 3xF; feat: kernel Sx3), K split over blockIdx.z ----------
// xhat NHWC bf16, wfrag [3][256][2][4][32][16]; per-pos 24KB weight slice staged in LDS,
// partial out [8][B][C][S] f32 (no bias)
__global__ __launch_bounds__(256) void k_widec(const unsigned short* __restrict__ xhat,
                                               const unsigned short* __restrict__ wfrag,
                                               float* __restrict__ outp, int mode) {
  __shared__ unsigned short wlds[12288];              // 3 taps x 2 kc chunks x 2048 = 24 KB
  const int t = threadIdx.x;
  const int lane = t & 31;
  const int wave = t >> 5;
  const int mt = blockIdx.x * 8 + wave;               // 0..15 m-tiles
  const int b = blockIdx.y;
  const int z = blockIdx.z;                           // K slice: pos in [z*32, z*32+32)
  const int m = lane & 15, half = lane >> 4;
  const int s = mt * 16 + m;
  const v4u z4 = {0u, 0u, 0u, 0u};
  v8f acc0 = {}, acc1 = {}, acc2 = {}, acc3 = {};

  for (int pos = z * 32; pos < z * 32 + 32; ++pos) {
    // stage this pos's weight fragments (all taps, kc, nc) cooperatively
    __syncthreads();
    for (int u = 0; u < 3; ++u) {
      const int linear = u * 4096 + t * 16;           // [tap][rest-of-2-chunks]
      const int tap = linear >> 12;
      const int rest = linear & 4095;
      const size_t gsrc = (size_t)((tap * 256 + pos) * 2) * 2048 + rest;
      *(v4u*)(wlds + linear)     = *(const v4u*)(wfrag + gsrc);
      *(v4u*)(wlds + linear + 8) = *(const v4u*)(wfrag + gsrc + 8);
    }
    __syncthreads();

    // prefetch next pos's center-tap activation row into caches (global_prefetch_b8)
    if (pos + 1 < z * 32 + 32) {
      const size_t nb = (mode == 0)
          ? (size_t)((b * 256 + s) * 256 + (pos + 1)) * 64
          : (size_t)((b * 256 + (pos + 1)) * 256 + s) * 64;
      __builtin_prefetch((const void*)(xhat + nb), 0, 1);
    }

    for (int tap = 0; tap < 3; ++tap) {
      const int sm = s + tap - 1;
      const bool inb = (sm >= 0) && (sm < 256);
      const int smc = inb ? sm : 0;
      size_t abase;
      if (mode == 0) abase = (size_t)((b * 256 + smc) * 256 + pos) * 64;  // seq: row pad
      else           abase = (size_t)((b * 256 + pos) * 256 + smc) * 64;  // feat: col pad
      for (int kc = 0; kc < 2; ++kc) {
        ABu a;
        if (inb) {
          a.q[0] = *(const v4u*)(xhat + abase + kc * 32 + half * 16);
          a.q[1] = *(const v4u*)(xhat + abase + kc * 32 + half * 16 + 8);
        } else { a.q[0] = z4; a.q[1] = z4; }
        const unsigned short* wb = wlds + tap * 4096 + kc * 2048 + lane * 16;
        ABu b0, b1, b2, b3;
        b0.q[0] = *(const v4u*)(wb);        b0.q[1] = *(const v4u*)(wb + 8);
        b1.q[0] = *(const v4u*)(wb + 512);  b1.q[1] = *(const v4u*)(wb + 520);
        b2.q[0] = *(const v4u*)(wb + 1024); b2.q[1] = *(const v4u*)(wb + 1032);
        b3.q[0] = *(const v4u*)(wb + 1536); b3.q[1] = *(const v4u*)(wb + 1544);
        acc0 = __builtin_amdgcn_wmma_f32_16x16x32_bf16(false, a.v, false, b0.v, (short)0, acc0, false, false);
        acc1 = __builtin_amdgcn_wmma_f32_16x16x32_bf16(false, a.v, false, b1.v, (short)0, acc1, false, false);
        acc2 = __builtin_amdgcn_wmma_f32_16x16x32_bf16(false, a.v, false, b2.v, (short)0, acc2, false, false);
        acc3 = __builtin_amdgcn_wmma_f32_16x16x32_bf16(false, a.v, false, b3.v, (short)0, acc3, false, false);
      }
    }
  }
  const int n = lane & 15;
  v8f accs[4] = {acc0, acc1, acc2, acc3};
  for (int nc = 0; nc < 4; ++nc) {
    float* oc = outp + (size_t)z * 131072 + (size_t)(b * 64 + nc * 16 + n) * 256;
    for (int r = 0; r < 8; ++r) oc[mt * 16 + r + 8 * half] = accs[nc][r];
  }
}

__global__ __launch_bounds__(256) void k_reduce_wide(const float* __restrict__ part,
                                                     const float* __restrict__ bias,
                                                     float* __restrict__ outq) {
  const int idx = blockIdx.x * 256 + threadIdx.x;     // 512 blocks -> 131072
  const int c = (idx >> 8) & 63;
  float s = bias[c];
  for (int zz = 0; zz < 8; ++zz) s += part[(size_t)zz * 131072 + idx];
  outq[idx] = s;
}

// ---------- softmax over last dim (256) in place ----------
__global__ __launch_bounds__(256) void k_softmax256(float* __restrict__ q) {
  const int row = blockIdx.x;
  const int t = threadIdx.x;
  float v = q[(size_t)row * 256 + t];
  __shared__ float rs[256];
  rs[t] = v; __syncthreads();
  for (int off = 128; off > 0; off >>= 1) { if (t < off) rs[t] = fmaxf(rs[t], rs[t + off]); __syncthreads(); }
  const float mx = rs[0];
  __syncthreads();
  const float e = __expf(v - mx);
  rs[t] = e; __syncthreads();
  for (int off = 128; off > 0; off >>= 1) { if (t < off) rs[t] += rs[t + off]; __syncthreads(); }
  q[(size_t)row * 256 + t] = e / rs[0];
}

// ---------- final rank-1 combination ----------
// out[s,f] = (1-a)*( q1[s]*r1[f] + (x.q2[s] + q1[s]*d)*k2[f] ) + a*x[s,f]
__global__ __launch_bounds__(256) void k_final(const float* __restrict__ xn,
    const float* __restrict__ q1, const float* __restrict__ k1,
    const float* __restrict__ q2, const float* __restrict__ k2,
    const float* __restrict__ alpha_p, float* __restrict__ out) {
  const int bc = blockIdx.x;                          // 512 = B*C
  const int t = threadIdx.x;                          // f = t
  const float alpha = alpha_p[0];
  const float* xp = xn + (size_t)bc * 65536;
  float* op = out + (size_t)bc * 65536;
  __shared__ float sq1[256], sk1[256], sq2[256], sk2[256], sxq2[256], wpart[2048], dpart[8];
  sq1[t] = q1[(size_t)bc * 256 + t];
  sk1[t] = k1[(size_t)bc * 256 + t];
  sq2[t] = q2[(size_t)bc * 256 + t];
  sk2[t] = k2[(size_t)bc * 256 + t];
  __syncthreads();
  const int wv = t >> 5, lane = t & 31;
  const float q2t = sq2[t];
  float r1 = 0.f;
  for (int s = 0; s < 256; ++s) {
    const float xv = xp[(size_t)s * 256 + t];
    r1 += sk1[s] * xv;
    float p = xv * q2t;
    for (int off = 16; off > 0; off >>= 1) p += __shfl_down(p, off, 32);
    if (lane == 0) wpart[wv * 256 + s] = p;
  }
  float pd = r1 * q2t;
  for (int off = 16; off > 0; off >>= 1) pd += __shfl_down(pd, off, 32);
  if (lane == 0) dpart[wv] = pd;
  __syncthreads();
  {
    float sxv = 0.f;
    for (int w = 0; w < 8; ++w) sxv += wpart[w * 256 + t];
    sxq2[t] = sxv;
  }
  float d = 0.f;
  for (int w = 0; w < 8; ++w) d += dpart[w];
  __syncthreads();
  const float k2t = sk2[t];
  const float oma = 1.f - alpha;
  for (int s = 0; s < 256; ++s) {
    const float coef = sxq2[s] + sq1[s] * d;
    const float o = sq1[s] * r1 + coef * k2t;
    op[(size_t)s * 256 + t] = oma * o + alpha * xp[(size_t)s * 256 + t];
  }
}

// ---------- host ----------
extern "C" void kernel_launch(void* const* d_in, const int* in_sizes, int n_in,
                              void* d_out, int out_size, void* d_ws, size_t ws_size,
                              hipStream_t stream) {
  (void)in_sizes; (void)n_in; (void)out_size; (void)ws_size;
  const float* x      = (const float*)d_in[0];
  const float* res_g1 = (const float*)d_in[1];
  const float* res_b1 = (const float*)d_in[2];
  const float* res_w1 = (const float*)d_in[3];
  const float* res_c1 = (const float*)d_in[4];
  const float* res_g2 = (const float*)d_in[5];
  const float* res_b2 = (const float*)d_in[6];
  const float* res_w2 = (const float*)d_in[7];
  const float* res_c2 = (const float*)d_in[8];
  const float* sq_g = (const float*)d_in[9];
  const float* sq_b = (const float*)d_in[10];
  const float* sq_w = (const float*)d_in[11];
  const float* sq_c = (const float*)d_in[12];
  const float* sk_g = (const float*)d_in[13];
  const float* sk_b = (const float*)d_in[14];
  const float* sk_w = (const float*)d_in[15];
  const float* sk_c = (const float*)d_in[16];
  const float* fq_g = (const float*)d_in[17];
  const float* fq_b = (const float*)d_in[18];
  const float* fq_w = (const float*)d_in[19];
  const float* fq_c = (const float*)d_in[20];
  const float* fk_g = (const float*)d_in[21];
  const float* fk_b = (const float*)d_in[22];
  const float* fk_w = (const float*)d_in[23];
  const float* fk_c = (const float*)d_in[24];
  const float* alpha = (const float*)d_in[25];
  float* out = (float*)d_out;

  char* base = (char*)d_ws;
  size_t off = 0;
  auto take = [&](size_t bytes) -> void* {
    void* p = base + off;
    off += (bytes + 255) & ~(size_t)255;
    return p;
  };
  unsigned short* bufBF = (unsigned short*)take(67108864);   // a1 -> a2 -> xhat (NHWC bf16)
  float* bufA  = (float*)take(134217728);                    // y1 -> v2 (NHWC f32)
  float* xnew  = (float*)take(134217728);                    // x + v2 (NCHW f32)
  unsigned short* wf1  = (unsigned short*)take(73728);
  unsigned short* wf2  = (unsigned short*)take(73728);
  unsigned short* wfSQ = (unsigned short*)take(6291456);
  unsigned short* wfSK = (unsigned short*)take(6291456);
  unsigned short* wfFQ = (unsigned short*)take(6291456);
  unsigned short* wfFK = (unsigned short*)take(6291456);
  float* biasSQ = (float*)take(256);
  float* biasSK = (float*)take(256);
  float* biasFQ = (float*)take(256);
  float* biasFK = (float*)take(256);
  float* mean1 = (float*)take(256);
  float* rstd1 = (float*)take(256);
  float* mean2 = (float*)take(256);
  float* rstd2 = (float*)take(256);
  float* mean3 = (float*)take(256);
  float* rstd3 = (float*)take(256);
  float* part  = (float*)take(131072);
  float* partW = (float*)take(4194304);                      // 8 K-slices x [B][C][S]
  float* q1b   = (float*)take(524288);
  float* k1b   = (float*)take(524288);
  float* q2b   = (float*)take(524288);
  float* k2b   = (float*)take(524288);

  // weight packing (data independent)
  k_pack_res<<<144, 256, 0, stream>>>(res_w1, wf1);
  k_pack_res<<<144, 256, 0, stream>>>(res_w2, wf2);
  k_pack_wide<<<12288, 256, 0, stream>>>(sq_w, sq_g, wfSQ, 0);
  k_pack_wide<<<12288, 256, 0, stream>>>(sk_w, sk_g, wfSK, 0);
  k_pack_wide<<<12288, 256, 0, stream>>>(fq_w, fq_g, wfFQ, 1);
  k_pack_wide<<<12288, 256, 0, stream>>>(fk_w, fk_g, wfFK, 1);
  k_fold_bias<<<64, 256, 0, stream>>>(sq_w, sq_b, sq_c, biasSQ);
  k_fold_bias<<<64, 256, 0, stream>>>(sk_w, sk_b, sk_c, biasSK);
  k_fold_bias<<<64, 256, 0, stream>>>(fq_w, fq_b, fq_c, biasFQ);
  k_fold_bias<<<64, 256, 0, stream>>>(fk_w, fk_b, fk_c, biasFK);

  // ResNet layer
  k_stats_nchw<<<64, 256, 0, stream>>>(x, mean1, rstd1);
  k_bn_nchw2nhwc<<<dim3(256, 8), 256, 0, stream>>>(x, mean1, rstd1, res_g1, res_b1, bufBF, 1);
  k_conv3x3<<<dim3(2, 256, 8), 256, 73728, stream>>>(bufBF, wf1, res_c1, bufA);
  k_stats_nhwc_part<<<256, 256, 0, stream>>>(bufA, part);
  k_stats_nhwc_fin<<<1, 64, 0, stream>>>(part, mean2, rstd2);
  k_bnrelu_nhwc<<<8192, 256, 0, stream>>>(bufA, mean2, rstd2, res_g2, res_b2, bufBF);
  k_conv3x3<<<dim3(2, 256, 8), 256, 73728, stream>>>(bufBF, wf2, res_c2, bufA);
  k_resadd<<<dim3(256, 8), 256, 0, stream>>>(x, bufA, xnew);

  // attention branches: shared BN normalize (gamma folded into packed weights, beta into bias)
  k_stats_nchw<<<64, 256, 0, stream>>>(xnew, mean3, rstd3);
  k_bn_nchw2nhwc<<<dim3(256, 8), 256, 0, stream>>>(xnew, mean3, rstd3, res_g1, res_b1, bufBF, 0);

  k_widec<<<dim3(2, 8, 8), 256, 0, stream>>>(bufBF, wfSQ, partW, 0);
  k_reduce_wide<<<512, 256, 0, stream>>>(partW, biasSQ, q1b);
  k_widec<<<dim3(2, 8, 8), 256, 0, stream>>>(bufBF, wfSK, partW, 0);
  k_reduce_wide<<<512, 256, 0, stream>>>(partW, biasSK, k1b);
  k_widec<<<dim3(2, 8, 8), 256, 0, stream>>>(bufBF, wfFQ, partW, 1);
  k_reduce_wide<<<512, 256, 0, stream>>>(partW, biasFQ, q2b);
  k_widec<<<dim3(2, 8, 8), 256, 0, stream>>>(bufBF, wfFK, partW, 1);
  k_reduce_wide<<<512, 256, 0, stream>>>(partW, biasFK, k2b);

  k_softmax256<<<512, 256, 0, stream>>>(q1b);
  k_softmax256<<<512, 256, 0, stream>>>(k1b);
  k_softmax256<<<512, 256, 0, stream>>>(q2b);
  k_softmax256<<<512, 256, 0, stream>>>(k2b);

  k_final<<<512, 256, 0, stream>>>(xnew, q1b, k1b, q2b, k2b, alpha, out);
}
